// EdgeAttrs_75453985456536
// MI455X (gfx1250) — compile-verified, matrix operand order checked by reading, not run on hardware
//
#include <hip/hip_runtime.h>
#include <hip/hip_bf16.h>
#include <math.h>

typedef __attribute__((ext_vector_type(16))) _Float16 v16h;
typedef __attribute__((ext_vector_type(8)))  _Float16 v8h;
typedef __attribute__((ext_vector_type(8)))  float    v8f;
typedef __attribute__((ext_vector_type(4)))  float    v4f;

// ---------------------------------------------------------------------------
// CDNA5 WMMA f32 = f16 x f16 + f32, 16x16x32, wave32 (ISA 7.12.2 layouts).
// A frag: lane m = lane&15, element e -> K = (e&7) + 8*(lane>>4) + 16*(e>>3).
// B frag mirrors A with n = lane&15. C/D: vgpr r -> row r + 8*(lane>>4).
// Packed-B layout: fragment f is 512 halves; lane L owns halves [L*16, L*16+16).
// ---------------------------------------------------------------------------
#define FRAG_HALVES 512
#define FRAG_BYTES  1024

__device__ __forceinline__ int frag_k(int e, int half) {
  return (e & 7) + 8 * half + ((e >> 3) << 4);
}

// Build an A fragment from an f16 row (contiguous 2x16B loads).
__device__ __forceinline__ v16h load_a16(const _Float16* __restrict__ row,
                                         int kb, int half) {
  v8h lo = *(const v8h*)(row + kb + 8 * half);
  v8h hi = *(const v8h*)(row + kb + 16 + 8 * half);
  return __builtin_shufflevector(lo, hi, 0, 1, 2, 3, 4, 5, 6, 7,
                                 8, 9, 10, 11, 12, 13, 14, 15);
}

// ---------------------------------------------------------------------------
// pack_b: one 32-thread block per fragment (ks*4 + t). src row-major [K,64] f32.
// ---------------------------------------------------------------------------
__global__ void pack_b_kernel(const float* __restrict__ src,
                              _Float16* __restrict__ dst) {
  const int frag = blockIdx.x;
  const int ks = frag >> 2, t = frag & 3;
  const int lane = threadIdx.x & 31;
  const int half = lane >> 4, n = lane & 15;
  v16h b;
#pragma unroll
  for (int e = 0; e < 16; ++e)
    b[e] = (_Float16)src[(ks * 32 + frag_k(e, half)) * 64 + t * 16 + n];
  *(v16h*)(dst + frag * FRAG_HALVES + lane * 16) = b;
}

// ---------------------------------------------------------------------------
// norm + f16 copy of x
// ---------------------------------------------------------------------------
__global__ void norm_x16_kernel(const float* __restrict__ x,
                                float* __restrict__ nrm,
                                _Float16* __restrict__ x16, int N) {
  const int i = blockIdx.x * blockDim.x + threadIdx.x;
  if (i >= N) return;
  const float* xp = x + (long)i * 128;
  _Float16* op = x16 + (long)i * 128;
  float s = 0.f;
#pragma unroll 8
  for (int d = 0; d < 128; ++d) {
    const float v = xp[d];
    s = fmaf(v, v, s);
    op[d] = (_Float16)v;
  }
  nrm[i] = fmaxf(sqrtf(s), 1e-8f);
}

// ---------------------------------------------------------------------------
// proj: z = relu(x @ W + b); x16:[N,128] f16, Bp: 16 packed frags (K=128)
// ---------------------------------------------------------------------------
__global__ void proj_kernel(const _Float16* __restrict__ x16,
                            const _Float16* __restrict__ Bp,
                            const float* __restrict__ bias,
                            _Float16* __restrict__ z, int N) {
  __shared__ v16h sB[16 * 32];
  {
    const v16h* gB = (const v16h*)Bp;
    for (int i = threadIdx.x; i < 16 * 32; i += blockDim.x) sB[i] = gB[i];
  }
  __syncthreads();

  const int lane = threadIdx.x & 31, wave = threadIdx.x >> 5;
  const int half = lane >> 4, n = lane & 15;
  const int rowBase = (blockIdx.x * 8 + wave) * 16;
  if (rowBase >= N) return;
  int myRow = rowBase + (lane & 15);
  if (myRow >= N) myRow = N - 1;
  const _Float16* xrow = x16 + (long)myRow * 128;

  v8f c[4] = {};
#pragma unroll
  for (int ks = 0; ks < 4; ++ks) {
    const v16h a = load_a16(xrow, ks * 32, half);
#pragma unroll
    for (int t = 0; t < 4; ++t) {
      const v16h b = sB[(ks * 4 + t) * 32 + lane];
      c[t] = __builtin_amdgcn_wmma_f32_16x16x32_f16(false, a, false, b,
                                                    (short)0, c[t], false, false);
    }
  }
#pragma unroll
  for (int t = 0; t < 4; ++t) {
    const float bv = bias[t * 16 + n];
#pragma unroll
    for (int r = 0; r < 8; ++r) {
      const int m = rowBase + r + 8 * half;
      float v = c[t][r] + bv;
      v = v > 0.f ? v : 0.f;
      if (m < N) z[(long)m * 64 + t * 16 + n] = (_Float16)v;
    }
  }
}

// ---------------------------------------------------------------------------
// compose: u = z @ We_sub; z:[N,64] f16, Bp: 8 packed frags (K=64), u f32
// ---------------------------------------------------------------------------
__global__ void compose_kernel(const _Float16* __restrict__ z,
                               const _Float16* __restrict__ Bp,
                               float* __restrict__ u, int N) {
  __shared__ v16h sB[8 * 32];
  {
    const v16h* gB = (const v16h*)Bp;
    for (int i = threadIdx.x; i < 8 * 32; i += blockDim.x) sB[i] = gB[i];
  }
  __syncthreads();

  const int lane = threadIdx.x & 31, wave = threadIdx.x >> 5;
  const int half = lane >> 4, n = lane & 15;
  const int rowBase = (blockIdx.x * 8 + wave) * 16;
  if (rowBase >= N) return;
  int myRow = rowBase + (lane & 15);
  if (myRow >= N) myRow = N - 1;
  const _Float16* zrow = z + (long)myRow * 64;

  v8f c[4] = {};
#pragma unroll
  for (int ks = 0; ks < 2; ++ks) {
    const v16h a = load_a16(zrow, ks * 32, half);
#pragma unroll
    for (int t = 0; t < 4; ++t) {
      const v16h b = sB[(ks * 4 + t) * 32 + lane];
      c[t] = __builtin_amdgcn_wmma_f32_16x16x32_f16(false, a, false, b,
                                                    (short)0, c[t], false, false);
    }
  }
#pragma unroll
  for (int t = 0; t < 4; ++t)
#pragma unroll
    for (int r = 0; r < 8; ++r) {
      const int m = rowBase + r + 8 * half;
      if (m < N) u[(long)m * 64 + t * 16 + n] = c[t][r];
    }
}

// ---------------------------------------------------------------------------
// edge kernel: one wave -> 16 edges. Bp: 16 packed frags, [W4 (8) | Wattr (8)].
// acc = u1[row]+u2[col]+u3[row]-u3[col] + s*We[256] + be  (gathered C frags)
// acc += sqrt(z4[row]*z4[col]) @ W4 + edge_attr @ Wattr   (WMMA)
// ---------------------------------------------------------------------------
__global__ void edge_kernel(const float* __restrict__ x,
                            const int* __restrict__ ei,      // [2,E] int32
                            const float* __restrict__ ea,    // [E,64]
                            const _Float16* __restrict__ z4, // [N,64]
                            const float* __restrict__ u1,
                            const float* __restrict__ u2,
                            const float* __restrict__ u3,
                            const float* __restrict__ nrm,
                            const _Float16* __restrict__ Bp,
                            const float* __restrict__ Wcos,  // We row 256
                            const float* __restrict__ be,    // [64]
                            float* __restrict__ out, int E) {
  __shared__ v16h sB[16 * 32];
  {
    const v16h* gB = (const v16h*)Bp;
    for (int i = threadIdx.x; i < 16 * 32; i += blockDim.x) sB[i] = gB[i];
  }
  __syncthreads();

  const int lane = threadIdx.x & 31, wave = threadIdx.x >> 5;
  const int half = lane >> 4, n = lane & 15;
  const int base = (blockIdx.x * 8 + wave) * 16;
  if (base >= E) return;

  // prefetch next block's slice of the streaming edge_attr operand
  {
    int pe = base + 128;
    pe = pe < E ? pe : E - 1;
    __builtin_prefetch(ea + (long)pe * 64 + lane, 0, 1);
  }

  // --- cosine similarity for edge (base + lane&15); half-wave splits D=128 ---
  int me = base + (lane & 15);
  if (me >= E) me = E - 1;
  const int mr = ei[me], mc = ei[E + me];
  const float* xr = x + (long)mr * 128 + half * 64;
  const float* xc = x + (long)mc * 128 + half * 64;
  float dot = 0.f;
#pragma unroll 8
  for (int d = 0; d < 64; ++d) dot = fmaf(xr[d], xc[d], dot);
  dot += __shfl_xor(dot, 16, 32);
  const float s = dot / (nrm[mr] * nrm[mc]);

  // --- gather linear terms straight into C fragments ---
  v8f acc[4] = {};
#pragma unroll
  for (int r = 0; r < 8; ++r) {
    int em = base + r + 8 * half;
    if (em >= E) em = E - 1;
    const int re = ei[em], ce = ei[E + em];
    const float se = __shfl(s, r + 8 * half, 32);
    const float* u1p = u1 + (long)re * 64;
    const float* u2p = u2 + (long)ce * 64;
    const float* u3r = u3 + (long)re * 64;
    const float* u3c = u3 + (long)ce * 64;
#pragma unroll
    for (int t = 0; t < 4; ++t) {
      const int f = t * 16 + n;
      acc[t][r] = u1p[f] + u2p[f] + (u3r[f] - u3c[f]) + se * Wcos[f] + be[f];
    }
  }

  // --- A fragments: sqrt(z4[row]*z4[col]) and edge_attr, K = 64 (2 steps) ---
  v16h qa[2], aa[2];
  {
    const _Float16* zr = z4 + (long)mr * 64;
    const _Float16* zc = z4 + (long)mc * 64;
    const float* ep = ea + (long)me * 64;
#pragma unroll
    for (int ks = 0; ks < 2; ++ks) {
      const v16h ar = load_a16(zr, ks * 32, half);
      const v16h ac = load_a16(zc, ks * 32, half);
      const v4f e0 = *(const v4f*)(ep + ks * 32 + 8 * half);
      const v4f e1 = *(const v4f*)(ep + ks * 32 + 8 * half + 4);
      const v4f e2 = *(const v4f*)(ep + ks * 32 + 16 + 8 * half);
      const v4f e3 = *(const v4f*)(ep + ks * 32 + 16 + 8 * half + 4);
#pragma unroll
      for (int e = 0; e < 16; ++e) {
        const float p = (float)ar[e] * (float)ac[e];
        qa[ks][e] = (_Float16)sqrtf(p);
      }
#pragma unroll
      for (int e = 0; e < 4; ++e) {
        aa[ks][e]      = (_Float16)e0[e];
        aa[ks][e + 4]  = (_Float16)e1[e];
        aa[ks][e + 8]  = (_Float16)e2[e];
        aa[ks][e + 12] = (_Float16)e3[e];
      }
    }
  }

  // --- WMMA: acc += qa @ W4 + aa @ Wattr ---
#pragma unroll
  for (int t = 0; t < 4; ++t) {
#pragma unroll
    for (int ks = 0; ks < 2; ++ks) {
      const v16h b4 = sB[(ks * 4 + t) * 32 + lane];          // W4 frags 0..7
      const v16h ba = sB[(8 + ks * 4 + t) * 32 + lane];      // Wattr frags 8..15
      acc[t] = __builtin_amdgcn_wmma_f32_16x16x32_f16(false, qa[ks], false, b4,
                                                      (short)0, acc[t], false, false);
      acc[t] = __builtin_amdgcn_wmma_f32_16x16x32_f16(false, aa[ks], false, ba,
                                                      (short)0, acc[t], false, false);
    }
  }

  // --- ReLU + coalesced store ---
#pragma unroll
  for (int r = 0; r < 8; ++r) {
    const int em = base + r + 8 * half;
    if (em < E) {
#pragma unroll
      for (int t = 0; t < 4; ++t) {
        const float v = acc[t][r];
        out[(long)em * 64 + t * 16 + n] = v > 0.f ? v : 0.f;
      }
    }
  }
}

// ---------------------------------------------------------------------------
extern "C" void kernel_launch(void* const* d_in, const int* in_sizes, int n_in,
                              void* d_out, int out_size, void* d_ws, size_t ws_size,
                              hipStream_t stream) {
  const float* x  = (const float*)d_in[0];
  const int*   ei = (const int*)d_in[1];
  const float* ea = (const float*)d_in[2];
  const float* W[4]  = {(const float*)d_in[3], (const float*)d_in[5],
                        (const float*)d_in[7], (const float*)d_in[9]};
  const float* bb[4] = {(const float*)d_in[4], (const float*)d_in[6],
                        (const float*)d_in[8], (const float*)d_in[10]};
  const float* We = (const float*)d_in[11];
  const float* be = (const float*)d_in[12];

  const int N = in_sizes[0] / 128;
  const int E = in_sizes[2] / 64;

  // workspace carve-out (256B aligned)
  char* ws = (char*)d_ws;
  size_t off = 0;
  auto carve = [&](size_t bytes) -> void* {
    void* p = ws + off;
    off += (bytes + 255) & ~(size_t)255;
    return p;
  };
  _Float16* x16 = (_Float16*)carve((size_t)N * 128 * sizeof(_Float16));
  _Float16* z[4];
  for (int i = 0; i < 4; ++i) z[i] = (_Float16*)carve((size_t)N * 64 * sizeof(_Float16));
  float* u[3];
  for (int i = 0; i < 3; ++i) u[i] = (float*)carve((size_t)N * 64 * sizeof(float));
  float* nrm = (float*)carve((size_t)N * sizeof(float));
  _Float16* Wp[4];                                  // proj packs: 16 frags each
  for (int i = 0; i < 4; ++i) Wp[i] = (_Float16*)carve(16 * FRAG_BYTES);
  _Float16* Cp[3];                                  // compose packs: 8 frags each
  for (int i = 0; i < 3; ++i) Cp[i] = (_Float16*)carve(8 * FRAG_BYTES);
  _Float16* Ep = (_Float16*)carve(16 * FRAG_BYTES); // edge packs: W4 + Wattr

  // --- pack weights into WMMA-fragment layout (tiny, runs once per call) ---
  for (int i = 0; i < 4; ++i)
    pack_b_kernel<<<16, 32, 0, stream>>>(W[i], Wp[i]);            // K=128
  for (int i = 0; i < 3; ++i)
    pack_b_kernel<<<8, 32, 0, stream>>>(We + i * 64 * 64, Cp[i]); // K=64
  pack_b_kernel<<<8, 32, 0, stream>>>(We + 192 * 64, Ep);         // W4
  pack_b_kernel<<<8, 32, 0, stream>>>(We + 257 * 64, Ep + 8 * FRAG_HALVES); // Wattr

  norm_x16_kernel<<<(N + 255) / 256, 256, 0, stream>>>(x, nrm, x16, N);

  const int nodeBlocks = (N + 127) / 128;  // 8 waves/block * 16 rows/wave
  for (int i = 0; i < 4; ++i)
    proj_kernel<<<nodeBlocks, 256, 0, stream>>>(x16, Wp[i], bb[i], z[i], N);
  for (int i = 0; i < 3; ++i)
    compose_kernel<<<nodeBlocks, 256, 0, stream>>>(z[i], Cp[i], u[i], N);

  const int edgeBlocks = (E + 127) / 128;  // 8 waves/block * 16 edges/wave
  edge_kernel<<<edgeBlocks, 256, 0, stream>>>(x, ei, ea, z[3], u[0], u[1], u[2],
                                              nrm, Ep, We + 256 * 64, be,
                                              (float*)d_out, E);
}
